// MultiMonoidAttention_84293028151342
// MI455X (gfx1250) — matrix operand order repaired and checked by reading, not compiled, not run on hardware
//
#include <hip/hip_runtime.h>
#include <math.h>
#include <stdint.h>

// ---------------------------------------------------------------------------
// MultiMonoidAttention for MI455X (gfx1250, wave32, WMMA + async-LDS copies).
// Dims fixed by the reference: B=2, N=M=1024, D=1024, HD=256, MB=32.
// ---------------------------------------------------------------------------

#define BB   2
#define NN   1024
#define MM   1024
#define DD   1024
#define HD   256
#define MB_  32
#define LQP  192            // 6 stacked 32-wide head projections
#define SCALE_F 0.17677669529663687f   // 1/sqrt(32)

typedef __attribute__((ext_vector_type(16))) _Float16 v16h;
typedef __attribute__((ext_vector_type(8)))  float    v8f;

// ---------------------------------------------------------------------------
// Async global->LDS copy, 16 bytes per lane (ASYNCcnt-tracked, gfx1250).
// ---------------------------------------------------------------------------
__device__ __forceinline__ void async_b128(unsigned lds_byte, unsigned g_off,
                                           const void* base) {
  asm volatile("global_load_async_to_lds_b128 %0, %1, %2"
               :: "v"(lds_byte), "v"(g_off), "s"(base)
               : "memory");
}

// ---------------------------------------------------------------------------
// Double-buffered block-tiled WMMA GEMM:
//   C[rows,Ncols] = A[rows,K] * B^T (+ bias[col])
//   BKN=false : B stored [Ncols, K]  (weights, out_w)
//   BKN=true  : B stored [K, Ncols]  (attend: probs x V-projection)
// 128 threads = 4 waves; block computes a 64x64 C tile; wave w owns the
// 16-row strip [16w,16w+16) x 64 cols as 4 WMMA accumulators (A-frag reuse x4).
// Tiles are staged f32 in LDS by global_load_async_to_lds_b128; the copies for
// k-step i+1 run concurrently with the WMMAs of k-step i (each thread keeps
// exactly 8 copies per tile in flight; async loads complete in order, so
// s_wait_asynccnt 8 proves the current tile has landed).
// Requires: rows%64==0, Ncols%64==0, K%32==0 (guaranteed by the launcher).
// ---------------------------------------------------------------------------
template <bool BKN>
__global__ void __launch_bounds__(128)
mma_gemm_wmma_v3(const float* __restrict__ A, int lda,
                 const float* __restrict__ B, int ldb,
                 float* __restrict__ C, int ldc,
                 const float* __restrict__ bias, int K) {
  __shared__ float As[2][64 * 32];                  // [buf][row][k]   2x8 KB
  __shared__ float Bs[2][64 * 32];                  // layout per BKN, 2x8 KB
  const int lane = threadIdx.x;                     // 0..31 (wave32)
  const int wid  = threadIdx.y;                     // 0..3
  const int tid  = wid * 32 + lane;
  const int l16  = lane & 15;
  const int lhi  = lane >> 4;
  const int tn0  = blockIdx.x * 64;
  const int tm0  = blockIdx.y * 64;

  const v8f zero = {};
  v8f acc[4] = {zero, zero, zero, zero};

  const unsigned asb[2] = {(unsigned)(uintptr_t)(&As[0][0]),
                           (unsigned)(uintptr_t)(&As[1][0])};
  const unsigned bsb[2] = {(unsigned)(uintptr_t)(&Bs[0][0]),
                           (unsigned)(uintptr_t)(&Bs[1][0])};

  // Stage the 64x32 A tile and the B tile for k-offset k0 into buffer `buf`:
  // 8 async b128 copies per thread (4 A + 4 B).
  auto stage = [&](int buf, int k0) {
    const char* abase = (const char*)(A + (size_t)tm0 * lda + k0);
#pragma unroll
    for (int i = 0; i < 4; ++i) {
      const int c   = tid + i * 128;                // 0..511 16B-chunks
      const int row = c >> 3;
      const int k4  = (c & 7) << 2;
      async_b128(asb[buf] + (unsigned)((row * 32 + k4) * 4),
                 (unsigned)((row * lda + k4) * 4), abase);
    }
    const char* bbase = BKN ? (const char*)(B + (size_t)k0 * ldb + tn0)
                            : (const char*)(B + (size_t)tn0 * ldb + k0);
#pragma unroll
    for (int i = 0; i < 4; ++i) {
      const int c = tid + i * 128;
      if (BKN) {                                    // Bs[k][col], 32x64
        const int kk = c >> 4;
        const int c4 = (c & 15) << 2;
        async_b128(bsb[buf] + (unsigned)((kk * 64 + c4) * 4),
                   (unsigned)((kk * ldb + c4) * 4), bbase);
      } else {                                      // Bs[col][k], 64x32
        const int col = c >> 3;
        const int k4  = (c & 7) << 2;
        async_b128(bsb[buf] + (unsigned)((col * 32 + k4) * 4),
                   (unsigned)((col * ldb + k4) * 4), bbase);
      }
    }
  };

  const int nk = K >> 5;                            // K / 32 k-steps
  stage(0, 0);

  for (int ik = 0; ik < nk; ++ik) {
    const int cur = ik & 1;
    if (ik + 1 < nk) {
      // Prefetch next tile into the other buffer (read last at ik-1; its
      // trailing barrier already passed), then wait only for the current
      // tile's 8 copies: ASYNCcnt <= 8 leaves the prefetch in flight.
      stage(cur ^ 1, (ik + 1) << 5);
      asm volatile("s_wait_asynccnt 8" ::: "memory");
    } else {
      asm volatile("s_wait_asynccnt 0" ::: "memory");
    }
    __syncthreads();   // current tile visible to all waves

    // ---- A fragment (16x32 f16): row = l16, K = e+16*lhi ----
    v16h af;
    {
      const float* a0 = &As[cur][(wid * 16 + l16) * 32 + (lhi << 4)];
#pragma unroll
      for (int e = 0; e < 8; ++e) {
        af[e]     = (_Float16)a0[e];
        af[e + 8] = (_Float16)a0[e + 8];
      }
    }
    // ---- 4 B fragments + WMMA ----
#pragma unroll
    for (int j = 0; j < 4; ++j) {
      v16h bf;
      if (BKN) {
#pragma unroll
        for (int e = 0; e < 16; ++e)
          bf[e] = (_Float16)Bs[cur][(e + (lhi << 4)) * 64 + (j * 16 + l16)];
      } else {
        const float* bp = &Bs[cur][(j * 16 + l16) * 32 + (lhi << 4)];
#pragma unroll
        for (int e = 0; e < 16; ++e) bf[e] = (_Float16)bp[e];
      }
      acc[j] = __builtin_amdgcn_wmma_f32_16x16x32_f16(
          false, af, false, bf, (short)0, acc[j], false, false);
    }
    __syncthreads();   // protect buffer `cur` before it is restaged at ik+1
  }

  // ---- epilogue: C row = tm0 + 16*wid + r + 8*lhi, col = tn0 + 16j + l16 ----
#pragma unroll
  for (int j = 0; j < 4; ++j) {
    const int col = tn0 + j * 16 + l16;
    const float bv = bias ? bias[col] : 0.0f;
#pragma unroll
    for (int r = 0; r < 8; ++r)
      C[(size_t)(tm0 + wid * 16 + r + 8 * lhi) * ldc + col] = acc[j][r] + bv;
  }
}

// ---------------------------------------------------------------------------
// Pack 6 [32,1024] weight blocks + 6 [32] biases into [192,1024] / [192].
// ---------------------------------------------------------------------------
__global__ void mma_pack6(const float* w0, const float* w1, const float* w2,
                          const float* w3, const float* w4, const float* w5,
                          const float* b0, const float* b1, const float* b2,
                          const float* b3, const float* b4, const float* b5,
                          float* __restrict__ W, float* __restrict__ Bv) {
  const int idx = blockIdx.x * 256 + threadIdx.x;
  const float* ws[6] = {w0, w1, w2, w3, w4, w5};
  const float* bs[6] = {b0, b1, b2, b3, b4, b5};
  if (idx < LQP * DD) W[idx] = ws[idx >> 15][idx & 32767];
  if (idx < LQP)      Bv[idx] = bs[idx >> 5][idx & 31];
}

// ---------------------------------------------------------------------------
// Pack sign bits of a 32-wide column slice into one u32 per row.
// ---------------------------------------------------------------------------
__global__ void mma_bits_kernel(const float* __restrict__ P, int ldp, int col0,
                                unsigned* __restrict__ outbits, int nrows) {
  const int row = blockIdx.x * blockDim.x + threadIdx.x;
  if (row >= nrows) return;
  const float* __restrict__ p = P + (size_t)row * ldp + col0;
  unsigned m = 0u;
#pragma unroll
  for (int j = 0; j < 32; ++j)
    m |= (p[j] > 0.0f) ? (1u << j) : 0u;
  outbits[row] = m;
}

// ---------------------------------------------------------------------------
// Fused score kernel: all four [B,N,M] score tensors.
//   head0: boolean hamming (xnor-popcount)         * SCALE
//   head1: -tropical_minplus / softplus(t_temp)
//   head2: real 32-dot                             * SCALE
//   head3: softmax(p_fusion) . (bool, trop, real)  * SCALE
// ---------------------------------------------------------------------------
__global__ void mma_scores_kernel(const float* __restrict__ QP,
                                  const float* __restrict__ KP,
                                  const unsigned* __restrict__ qb_b,
                                  const unsigned* __restrict__ kb_b,
                                  const unsigned* __restrict__ qb_p,
                                  const unsigned* __restrict__ kb_p,
                                  const float* __restrict__ t_temp,
                                  const float* __restrict__ p_fusion,
                                  float* __restrict__ S) {
  const int b  = blockIdx.z;
  const int n0 = blockIdx.y * 16;
  const int k0 = blockIdx.x * 16;
  const int tx = threadIdx.x, ty = threadIdx.y;

  __shared__ float q_s[16][161];   // cols 32..191 of QP (padded)
  __shared__ float k_s[16][161];

  const int tid = ty * 16 + tx;
  for (int i = tid; i < 16 * 160; i += 256) {
    const int r = i / 160, c = i % 160;
    q_s[r][c] = QP[(size_t)(b * NN + n0 + r) * LQP + 32 + c];
    k_s[r][c] = KP[(size_t)(b * MM + k0 + r) * LQP + 32 + c];
  }
  __syncthreads();

  const int qrow = b * NN + n0 + ty;
  const int krow = b * MM + k0 + tx;

  const float sim_b0 = (float)__popc(~(qb_b[qrow] ^ kb_b[krow])) * (1.0f / 32.0f);
  const float sim_bp = (float)__popc(~(qb_p[qrow] ^ kb_p[krow])) * (1.0f / 32.0f);

  // local cols: tq=0..31, rq=32..63, pq1(trop)=96..127, pq2(real)=128..159
  float tmin = INFINITY, tminp = INFINITY, dot_r = 0.0f, dot_p = 0.0f;
#pragma unroll
  for (int m = 0; m < 32; ++m) {
    tmin   = fminf(tmin,  q_s[ty][m]       + k_s[tx][m]);
    dot_r += q_s[ty][32 + m]  * k_s[tx][32 + m];
    tminp  = fminf(tminp, q_s[ty][96 + m]  + k_s[tx][96 + m]);
    dot_p += q_s[ty][128 + m] * k_s[tx][128 + m];
  }

  const float sp = log1pf(__expf(t_temp[0]));   // softplus
  const float f0 = p_fusion[0], f1 = p_fusion[1], f2 = p_fusion[2];
  const float fm = fmaxf(f0, fmaxf(f1, f2));
  const float e0 = __expf(f0 - fm), e1 = __expf(f1 - fm), e2 = __expf(f2 - fm);
  const float inv_es = 1.0f / (e0 + e1 + e2);

  const size_t idx = ((size_t)b * NN + n0 + ty) * MM + (k0 + tx);
  const size_t HS  = (size_t)BB * NN * MM;
  S[idx]          = sim_b0 * SCALE_F;
  S[HS + idx]     = -tmin / sp;
  S[2 * HS + idx] = dot_r * SCALE_F;
  S[3 * HS + idx] = ((e0 * sim_bp + e1 * (-tminp) + e2 * dot_p) * inv_es) * SCALE_F;
}

// ---------------------------------------------------------------------------
// In-place row softmax over M; one 256-thread block per row.
// ---------------------------------------------------------------------------
__global__ void mma_softmax_rows(float* __restrict__ S, int M) {
  float* __restrict__ row = S + (size_t)blockIdx.x * M;
  __shared__ float red[256];
  const int t = threadIdx.x;

  float mx = -INFINITY;
  for (int i = t; i < M; i += 256) mx = fmaxf(mx, row[i]);
  red[t] = mx; __syncthreads();
  for (int s = 128; s > 0; s >>= 1) {
    if (t < s) red[t] = fmaxf(red[t], red[t + s]);
    __syncthreads();
  }
  mx = red[0]; __syncthreads();

  float sum = 0.0f;
  for (int i = t; i < M; i += 256) {
    const float e = __expf(row[i] - mx);
    row[i] = e;
    sum += e;
  }
  red[t] = sum; __syncthreads();
  for (int s = 128; s > 0; s >>= 1) {
    if (t < s) red[t] += red[t + s];
    __syncthreads();
  }
  const float inv = 1.0f / red[0];
  for (int i = t; i < M; i += 256) row[i] *= inv;
}

// ---------------------------------------------------------------------------
// Host-side launcher
// ---------------------------------------------------------------------------
static void launch_gemm(const float* A, int lda, const float* B, int ldb,
                        bool bkn, float* C, int ldc, const float* bias,
                        int K, int rows, int ncols, hipStream_t s) {
  dim3 blk(32, 4, 1);
  dim3 grd((unsigned)(ncols / 64), (unsigned)(rows / 64), 1);
  if (bkn)
    mma_gemm_wmma_v3<true><<<grd, blk, 0, s>>>(A, lda, B, ldb, C, ldc, bias, K);
  else
    mma_gemm_wmma_v3<false><<<grd, blk, 0, s>>>(A, lda, B, ldb, C, ldc, bias, K);
}

extern "C" void kernel_launch(void* const* d_in, const int* in_sizes, int n_in,
                              void* d_out, int out_size, void* d_ws, size_t ws_size,
                              hipStream_t stream) {
  (void)in_sizes; (void)n_in; (void)out_size; (void)ws_size;
  const float* Q      = (const float*)d_in[0];
  const float* Kin    = (const float*)d_in[1];
  const float* V      = (const float*)d_in[2];
  const float* bq_w   = (const float*)d_in[3];
  const float* bq_b   = (const float*)d_in[4];
  const float* bk_w   = (const float*)d_in[5];
  const float* bk_b   = (const float*)d_in[6];
  const float* b_v    = (const float*)d_in[7];
  const float* tq_w   = (const float*)d_in[8];
  const float* tq_b   = (const float*)d_in[9];
  const float* tk_w   = (const float*)d_in[10];
  const float* tk_b   = (const float*)d_in[11];
  const float* t_v    = (const float*)d_in[12];
  const float* t_temp = (const float*)d_in[13];
  const float* rq_w   = (const float*)d_in[14];
  const float* rq_b   = (const float*)d_in[15];
  const float* rk_w   = (const float*)d_in[16];
  const float* rk_b   = (const float*)d_in[17];
  const float* r_v    = (const float*)d_in[18];
  const float* pq_w   = (const float*)d_in[19];
  const float* pq_b   = (const float*)d_in[20];
  const float* pk_w   = (const float*)d_in[21];
  const float* pk_b   = (const float*)d_in[22];
  const float* p_v    = (const float*)d_in[23];
  const float* p_fus  = (const float*)d_in[24];
  const float* out_w  = (const float*)d_in[25];
  const float* out_b  = (const float*)d_in[26];
  float* out = (float*)d_out;

  // ---- workspace layout ----
  const size_t BN = (size_t)BB * NN;             // 2048 rows
  char* ws = (char*)d_ws;
  size_t off = 0;
  float*    WQ   = (float*)(ws + off);  off += (size_t)LQP * DD * sizeof(float);
  float*    BQ   = (float*)(ws + off);  off += 256 * sizeof(float);
  float*    WK   = (float*)(ws + off);  off += (size_t)LQP * DD * sizeof(float);
  float*    BK   = (float*)(ws + off);  off += 256 * sizeof(float);
  float*    QP   = (float*)(ws + off);  off += BN * LQP * sizeof(float);
  float*    KP   = (float*)(ws + off);  off += BN * LQP * sizeof(float);
  float*    VP   = (float*)(ws + off);  off += BN * (4 * HD) * sizeof(float);
  unsigned* QB_b = (unsigned*)(ws + off); off += BN * sizeof(unsigned);
  unsigned* KB_b = (unsigned*)(ws + off); off += BN * sizeof(unsigned);
  unsigned* QB_p = (unsigned*)(ws + off); off += BN * sizeof(unsigned);
  unsigned* KB_p = (unsigned*)(ws + off); off += BN * sizeof(unsigned);
  float*    Sbuf = (float*)(ws + off);  off += (size_t)4 * BB * NN * MM * sizeof(float);
  float*    CAT  = (float*)(ws + off);  off += BN * (4 * HD) * sizeof(float);

  // ---- 0) pack the 6 Q-side / K-side projections into [192,1024] + [192] ----
  {
    dim3 blk(256), grd((unsigned)((LQP * DD + 255) / 256));
    mma_pack6<<<grd, blk, 0, stream>>>(
        bq_w, tq_w, rq_w, pq_w, pq_w + MB_ * DD, pq_w + 2 * MB_ * DD,
        bq_b, tq_b, rq_b, pq_b, pq_b + MB_,      pq_b + 2 * MB_, WQ, BQ);
    mma_pack6<<<grd, blk, 0, stream>>>(
        bk_w, tk_w, rk_w, pk_w, pk_w + MB_ * DD, pk_w + 2 * MB_ * DD,
        bk_b, tk_b, rk_b, pk_b, pk_b + MB_,      pk_b + 2 * MB_, WK, BK);
  }

  // ---- 1) Q/K projections as single dense GEMMs: [2048,1024]x[192,1024]^T ----
  launch_gemm(Q,   DD, WQ, DD, false, QP, LQP, BQ, DD, (int)BN, LQP, stream);
  launch_gemm(Kin, DD, WK, DD, false, KP, LQP, BK, DD, (int)BN, LQP, stream);

  // ---- 2) V projections: 4 heads stacked into VP [2048, 4*256] ----
  const float* vw[4] = { b_v, t_v, r_v, p_v };
  for (int h = 0; h < 4; ++h)
    launch_gemm(V, DD, vw[h], DD, false, VP + h * HD, 4 * HD, nullptr,
                DD, (int)BN, HD, stream);

  // ---- 3) pack sign bits for the two boolean-style heads ----
  {
    dim3 blk(256), grd((unsigned)((BN + 255) / 256));
    mma_bits_kernel<<<grd, blk, 0, stream>>>(QP, LQP, 0,       QB_b, (int)BN);
    mma_bits_kernel<<<grd, blk, 0, stream>>>(KP, LQP, 0,       KB_b, (int)BN);
    mma_bits_kernel<<<grd, blk, 0, stream>>>(QP, LQP, 3 * MB_, QB_p, (int)BN);
    mma_bits_kernel<<<grd, blk, 0, stream>>>(KP, LQP, 3 * MB_, KB_p, (int)BN);
  }

  // ---- 4) fused scores for all 4 heads ----
  {
    dim3 blk(16, 16, 1), grd(MM / 16, NN / 16, BB);
    mma_scores_kernel<<<grd, blk, 0, stream>>>(QP, KP, QB_b, KB_b, QB_p, KB_p,
                                               t_temp, p_fus, Sbuf);
  }

  // ---- 5) softmax over M, in place (4 heads x B x N contiguous rows) ----
  mma_softmax_rows<<<dim3(4 * BB * NN), dim3(256), 0, stream>>>(Sbuf, MM);

  // ---- 6) attend: o_h = probs_h x VP_h  -> CAT[:, h*256:(h+1)*256] ----
  const size_t HS = (size_t)BB * NN * MM;
  for (int h = 0; h < 4; ++h)
    for (int b = 0; b < BB; ++b)
      launch_gemm(Sbuf + h * HS + (size_t)b * NN * MM, MM,
                  VP + (size_t)b * MM * (4 * HD) + h * HD, 4 * HD, true,
                  CAT + (size_t)b * NN * (4 * HD) + h * HD, 4 * HD, nullptr,
                  MM, NN, HD, stream);

  // ---- 7) output projection: out = CAT x out_w^T + out_b ----
  launch_gemm(CAT, 4 * HD, out_w, 4 * HD, false, out, DD, out_b,
              4 * HD, (int)BN, DD, stream);
}
